// GATWithSentenceEmbedding_18674517803271
// MI455X (gfx1250) — matrix operand order repaired and checked by reading, not compiled, not run on hardware
//
#include <hip/hip_runtime.h>
#include <hip/hip_bf16.h>

// Problem dims (fixed by reference)
#define NN   2048
#define FF   256
#define HCC  256
#define SS   512
#define HH   4
#define EE   8192
#define BB   768

typedef __attribute__((ext_vector_type(16))) _Float16 v16h;
typedef __attribute__((ext_vector_type(8)))  float    v8f;

// CDNA5 async global->LDS path (ASYNCcnt-tracked), with safe fallback.
#if defined(__HIP_DEVICE_COMPILE__) && \
    __has_builtin(__builtin_amdgcn_global_load_async_to_lds_b128) && \
    __has_builtin(__builtin_amdgcn_s_wait_asynccnt)
#define USE_ASYNC_LDS 1
#else
#define USE_ASYNC_LDS 0
#endif

#if USE_ASYNC_LDS
// builtin signature is (V4i addrspace(1)*, V4i addrspace(3)*, imm offset, imm cpol)
typedef int i4v __attribute__((vector_size(16)));
typedef __attribute__((address_space(1))) i4v* gptr_i4;
typedef __attribute__((address_space(3))) i4v* lptr_i4;
#endif

__device__ __forceinline__ void cp16_g2l(_Float16* l, const _Float16* g) {
#if USE_ASYNC_LDS
  __builtin_amdgcn_global_load_async_to_lds_b128((gptr_i4)(void*)g, (lptr_i4)(void*)l, 0, 0);
#else
  *(int4*)l = *(const int4*)g;   // ds_store_b128 via VGPR bounce
#endif
}

// ---------------------------------------------------------------------------
// Helpers
// ---------------------------------------------------------------------------
__device__ __forceinline__ float atomicMaxF(float* addr, float val) {
  // Ordered float max via int/uint monotonic mapping (no NaNs in this pipeline)
  if (val >= 0.f)
    return __int_as_float(atomicMax((int*)addr, __float_as_int(val)));
  else
    return __uint_as_float(atomicMin((unsigned int*)addr, __float_as_uint(val)));
}

__global__ void k_init_f32(float* __restrict__ p, float v, int n) {
  int i = blockIdx.x * blockDim.x + threadIdx.x;
  if (i < n) p[i] = v;
}

__global__ void k_f32_to_f16(const float* __restrict__ in, _Float16* __restrict__ out, int n) {
  int i = blockIdx.x * blockDim.x + threadIdx.x;
  if (i < n) out[i] = (_Float16)in[i];
}

// in: [rows, cols] row-major f32  ->  out: [cols, rows] row-major f16 (i.e. B^T)
__global__ void k_f32_to_f16_t(const float* __restrict__ in, _Float16* __restrict__ out,
                               int rows, int cols) {
  int i = blockIdx.x * blockDim.x + threadIdx.x;
  if (i >= rows * cols) return;
  int r = i / cols, c = i % cols;
  out[(size_t)c * rows + r] = (_Float16)in[i];
}

// ---------------------------------------------------------------------------
// Streaming GEMV  y[j] = relu(sum_i v[i]*W[i,j] + b[j])   (W row-major [nin,nout])
// Bandwidth-bound: coalesced column access + global_prefetch_b8 ahead of stream.
// ---------------------------------------------------------------------------
__global__ void k_gemv_relu(const float* __restrict__ v, const float* __restrict__ W,
                            const float* __restrict__ b, float* __restrict__ y,
                            int nin, int nout) {
  int j = blockIdx.x * blockDim.x + threadIdx.x;
  if (j >= nout) return;
  float acc = 0.f;
  for (int i = 0; i < nin; ++i) {
    if (i + 16 < nin) __builtin_prefetch(&W[(size_t)(i + 16) * nout + j], 0, 1);
    acc = fmaf(v[i], W[(size_t)i * nout + j], acc);
  }
  acc += b[j];
  y[j] = fmaxf(acc, 0.f);
}

// fc2: orig[j] = concat(g1,g2) @ fc2_w + b ; mask = (orig > 0) ; cnt += mask
__global__ void k_fc2(const float* __restrict__ g1, const float* __restrict__ g2,
                      const float* __restrict__ W, const float* __restrict__ b,
                      float* __restrict__ orig, int* __restrict__ mask, int* __restrict__ cnt) {
  int j = blockIdx.x * blockDim.x + threadIdx.x;
  if (j >= EE) return;
  float acc = 0.f;
  for (int i = 0; i < EE; ++i) {
    if (i + 16 < EE) __builtin_prefetch(&W[(size_t)(i + 16) * EE + j], 0, 1);
    acc = fmaf(g1[i], W[(size_t)i * EE + j], acc);
  }
  for (int i = 0; i < EE; ++i) {
    if (i + 16 < EE) __builtin_prefetch(&W[(size_t)(i + EE + 16) * EE + j], 0, 1);
    acc = fmaf(g2[i], W[(size_t)(i + EE) * EE + j], acc);
  }
  acc += b[j];
  orig[j] = acc;
  int m = acc > 0.f;            // sigmoid(acc) > 0.5  <=>  acc > 0
  mask[j] = m;
  if (m) atomicAdd(cnt, 1);
}

// ---------------------------------------------------------------------------
// WMMA GEMM: C[M,N] = A[M,K] * B[K,N], f16 in / f32 acc.
// A row-major f16; B pre-transposed (Bt[N,K] row-major f16).
// Block = 256 threads (8 waves) computing a 64x128 C tile:
//   waves arranged 4(M) x 2(N); each wave owns a 16x64 strip (4 WMMA tiles,
//   A fragment reused 4x). A(64x32) and B(128x32) K-slabs are staged in LDS
//   via GLOBAL_LOAD_ASYNC_TO_LDS_B128 (ASYNCcnt) when available, then read
//   back with ds_load_b128 in the ISA 7.12.2 fragment layouts:
//     A 16x32: lanes 0-15 -> M=lane, K = {0..7,16..23}; lanes 16-31 -> K+8
//     B 32x16: lanes 0-15 -> N=lane, K = {0..7,16..23}; lanes 16-31 -> K+8
//     C 16x16: VGPR r -> M = r + 8*(lane>=16), N = lane&15
// M % 64 == 0, N % 128 == 0, K % 32 == 0 at every call site -> EXEC all-1s.
// ---------------------------------------------------------------------------
#define MBK 64
#define NBK 128
#define KBK 32

__global__ __launch_bounds__(256) void k_gemm_wmma(
    const _Float16* __restrict__ A, const _Float16* __restrict__ Bt,
    float* __restrict__ C, int M, int N, int K) {
  __shared__ __align__(16) _Float16 shA[MBK * KBK];   // [64][32]
  __shared__ __align__(16) _Float16 shB[NBK * KBK];   // [128][32]
  const int tilesN = N / NBK;
  const int bm = (blockIdx.x / tilesN) * MBK;
  const int bn = (blockIdx.x % tilesN) * NBK;
  const int t    = threadIdx.x;
  const int wave = t >> 5;
  const int lane = t & 31;
  const int hi   = lane >> 4;
  const int l15  = lane & 15;
  const int wm   = wave >> 1;        // 0..3 : M sub-tile
  const int wn   = wave & 1;         // 0..1 : N sub-tile

  v8f acc[4] = {};

  const int ar_cp = t >> 2;              // A copy: 64 rows x 4 chunks
  const int ac_cp = (t & 3) * 8;         // 8 halves = 16B per chunk

  for (int k0 = 0; k0 < K; k0 += KBK) {
    __syncthreads();                     // previous iteration's LDS reads done
    cp16_g2l(&shA[ar_cp * KBK + ac_cp], &A[(size_t)(bm + ar_cp) * K + k0 + ac_cp]);
#pragma unroll
    for (int j = 0; j < 2; ++j) {        // B copy: 128 rows x 4 chunks = 512
      int cid = t + 256 * j;
      int br = cid >> 2, bc = (cid & 3) * 8;
      cp16_g2l(&shB[br * KBK + bc], &Bt[(size_t)(bn + br) * K + k0 + bc]);
    }
#if USE_ASYNC_LDS
    __builtin_amdgcn_s_wait_asynccnt(0);
#endif
    __syncthreads();

    v16h a;
    const _Float16* ar = &shA[(wm * 16 + l15) * KBK + (hi ? 8 : 0)];
#pragma unroll
    for (int i = 0; i < 8; ++i)  a[i] = ar[i];        // K = k0 + (0..7)  (+8 hi)
#pragma unroll
    for (int i = 8; i < 16; ++i) a[i] = ar[8 + i];    // K = k0 + (16..23)(+8 hi)
#pragma unroll
    for (int nt = 0; nt < 4; ++nt) {
      v16h b;
      const _Float16* br2 = &shB[(wn * 64 + nt * 16 + l15) * KBK + (hi ? 8 : 0)];
#pragma unroll
      for (int i = 0; i < 16; ++i) b[i] = br2[i < 8 ? i : 8 + i];     // K = k0 + {0..7,16..23} (+8 hi)
      acc[nt] = __builtin_amdgcn_wmma_f32_16x16x32_f16(false, a, false, b,
                                                       (short)0, acc[nt], false, false);
    }
  }
#pragma unroll
  for (int nt = 0; nt < 4; ++nt) {
    float* crow = C + (size_t)(bm + wm * 16 + (hi ? 8 : 0)) * N
                    + bn + wn * 64 + nt * 16 + l15;
#pragma unroll
    for (int r = 0; r < 8; ++r) crow[(size_t)r * N] = acc[nt][r];
  }
}

// fc1 epilogue: h0 = relu(C + b[col]) + sent[col]; also f16 copy for next GEMM
__global__ void k_fc1_epi(float* __restrict__ C, const float* __restrict__ b,
                          const float* __restrict__ sent, _Float16* __restrict__ h16, int n) {
  int i = blockIdx.x * blockDim.x + threadIdx.x;
  if (i >= n) return;
  int col = i % SS;
  float v = fmaxf(C[i] + b[col], 0.f) + sent[col];
  C[i] = v;
  h16[i] = (_Float16)v;
}

// ---------------------------------------------------------------------------
// GAT attention
// ---------------------------------------------------------------------------
// al_s[n,h] = <xp[n,h,:], a_src[h,:]>, al_d likewise
__global__ void k_alpha(const float* __restrict__ xp, const float* __restrict__ a_src,
                        const float* __restrict__ a_dst, float* __restrict__ al_s,
                        float* __restrict__ al_d, int H, int C) {
  int idx = blockIdx.x * blockDim.x + threadIdx.x;
  if (idx >= NN * H) return;
  int n = idx / H, h = idx % H;
  const float* xr = xp + (size_t)n * H * C + (size_t)h * C;
  float as = 0.f, ad = 0.f;
  for (int c = 0; c < C; ++c) { as = fmaf(xr[c], a_src[h * C + c], as);
                                ad = fmaf(xr[c], a_dst[h * C + c], ad); }
  al_s[idx] = as; al_d[idx] = ad;
}

// pass 1: segment max over dst (valid edges + self loops)
__global__ void k_att_max(const int* __restrict__ src, const int* __restrict__ dst,
                          const int* __restrict__ mask, const float* __restrict__ al_s,
                          const float* __restrict__ al_d, float* __restrict__ m, int H) {
  int e = blockIdx.x * blockDim.x + threadIdx.x;
  if (e >= EE + NN) return;
  int s, d;
  if (e < EE) { if (!mask[e]) return; s = src[e]; d = dst[e]; }
  else        { s = d = e - EE; }
  for (int h = 0; h < H; ++h) {
    float a = al_s[s * H + h] + al_d[d * H + h];
    a = a > 0.f ? a : 0.2f * a;                    // leaky_relu(0.2)
    atomicMaxF(&m[d * H + h], a);
  }
}

// pass 2: ex = exp(alpha - m[d]); denom[d,h] += ex; acc[d,h,c] += ex * xp[s,h,c]
__global__ void k_att_agg(const int* __restrict__ src, const int* __restrict__ dst,
                          const int* __restrict__ mask, const float* __restrict__ al_s,
                          const float* __restrict__ al_d, const float* __restrict__ m,
                          const float* __restrict__ xp, float* __restrict__ denom,
                          float* __restrict__ acc, int H, int C) {
  int e = blockIdx.x;
  int s, d;
  if (e < EE) { if (!mask[e]) return; s = src[e]; d = dst[e]; }
  else        { s = d = e - EE; }
  int HC = H * C;
  for (int i = threadIdx.x; i < HC; i += blockDim.x) {
    int h = i / C;
    float a = al_s[s * H + h] + al_d[d * H + h];
    a = a > 0.f ? a : 0.2f * a;
    float ex = __expf(a - m[d * H + h]);
    atomicAdd(&acc[(size_t)d * HC + i], ex * xp[(size_t)s * HC + i]);
    if ((i % C) == 0) atomicAdd(&denom[d * H + h], ex);
  }
}

// finalize: out = act(acc/denom + bias); optional ELU and optional f16 copy
__global__ void k_att_final(float* __restrict__ acc, const float* __restrict__ denom,
                            const float* __restrict__ bias, _Float16* __restrict__ h16,
                            int H, int C, int do_elu) {
  int idx = blockIdx.x * blockDim.x + threadIdx.x;
  int n = NN * H * C;
  if (idx >= n) return;
  int nh = idx / C;
  int hc = idx % (H * C);
  float v = acc[idx] / denom[nh] + bias[hc];
  if (do_elu) v = v > 0.f ? v : (__expf(v) - 1.f);
  acc[idx] = v;
  if (h16) h16[idx] = (_Float16)v;
}

// ---------------------------------------------------------------------------
// Edge features + masked BN + final score
// ---------------------------------------------------------------------------
__global__ void k_ef(const float* __restrict__ h2, const int* __restrict__ src,
                     const int* __restrict__ dst, _Float16* __restrict__ ef) {
  int e = blockIdx.x, c = threadIdx.x;                 // 8192 blocks x 256
  float xi = h2[(size_t)src[e] * FF + c];
  float xj = h2[(size_t)dst[e] * FF + c];
  size_t base = (size_t)e * (4 * FF);
  ef[base + c]            = (_Float16)xi;
  ef[base + FF + c]       = (_Float16)xj;
  ef[base + 2 * FF + c]   = (_Float16)fabsf(xi - xj);
  ef[base + 3 * FF + c]   = (_Float16)(xi * xj);
}

// partial sums for masked-BN statistics (coalesced: thread = column)
__global__ void k_bn_partial(const float* __restrict__ z, const int* __restrict__ mask,
                             float* __restrict__ sum, float* __restrict__ sq, int ncol) {
  int c = threadIdx.x;                                 // blockDim == ncol
  int e0 = blockIdx.x * 256;
  float a = 0.f, q = 0.f;
  for (int e = e0; e < e0 + 256; ++e) {
    if (mask[e]) { float v = z[(size_t)e * ncol + c]; a += v; q += v * v; }
  }
  atomicAdd(&sum[c], a);
  atomicAdd(&sq[c], q);
}

// apply masked BN + relu (in place), optional f16 copy for next GEMM
__global__ void k_bn_apply(float* __restrict__ z, const float* __restrict__ sum,
                           const float* __restrict__ sq, const int* __restrict__ cnt,
                           const float* __restrict__ g, const float* __restrict__ b,
                           _Float16* __restrict__ z16, int n, int ncol) {
  int i = blockIdx.x * blockDim.x + threadIdx.x;
  if (i >= n) return;
  int c = i % ncol;
  float cn  = fmaxf((float)(*cnt), 1.f);
  float mu  = sum[c] / cn;
  float var = sq[c] / cn - mu * mu;
  float is  = rsqrtf(var + 1e-5f);
  float v = fmaxf((z[i] - mu) * is * g[c] + b[c], 0.f);
  z[i] = v;
  if (z16) z16[i] = (_Float16)v;
}

// score = z2 @ w3 + b3 ; out = sigmoid(mask ? score : -2.5)
__global__ void k_mlp3(const float* __restrict__ z2, const float* __restrict__ w3,
                       const float* __restrict__ b3, const int* __restrict__ mask,
                       float* __restrict__ out) {
  int e = blockIdx.x * blockDim.x + threadIdx.x;
  if (e >= EE) return;
  float acc = b3[0];
  const float* zr = z2 + (size_t)e * (HCC / 2);
  for (int c = 0; c < HCC / 2; ++c) acc = fmaf(zr[c], w3[c], acc);
  float lg = mask[e] ? acc : -2.5f;
  out[e] = 1.f / (1.f + __expf(-lg));
}

// ---------------------------------------------------------------------------
// Host launch
// ---------------------------------------------------------------------------
static inline int ceil_div(int a, int b) { return (a + b - 1) / b; }

extern "C" void kernel_launch(void* const* d_in, const int* in_sizes, int n_in,
                              void* d_out, int out_size, void* d_ws, size_t ws_size,
                              hipStream_t stream) {
  (void)in_sizes; (void)n_in; (void)out_size; (void)ws_size;
  const float* x        = (const float*)d_in[0];
  const int*   ei       = (const int*)d_in[1];
  const float* sent_emb = (const float*)d_in[2];
  const float* elp      = (const float*)d_in[3];
  const float* fc0_w = (const float*)d_in[4],  *fc0_b = (const float*)d_in[5];
  const float* fcl_w = (const float*)d_in[6],  *fcl_b = (const float*)d_in[7];
  const float* fce_w = (const float*)d_in[8],  *fce_b = (const float*)d_in[9];
  const float* fc2_w = (const float*)d_in[10], *fc2_b = (const float*)d_in[11];
  const float* fc1_w = (const float*)d_in[12], *fc1_b = (const float*)d_in[13];
  const float* c1_W  = (const float*)d_in[14], *c1_as = (const float*)d_in[15];
  const float* c1_ad = (const float*)d_in[16], *c1_b  = (const float*)d_in[17];
  const float* c2_W  = (const float*)d_in[18], *c2_as = (const float*)d_in[19];
  const float* c2_ad = (const float*)d_in[20], *c2_b  = (const float*)d_in[21];
  const float* m1_w  = (const float*)d_in[22];
  const float* bn1_g = (const float*)d_in[24], *bn1_b = (const float*)d_in[25];
  const float* m2_w  = (const float*)d_in[26];
  const float* bn2_g = (const float*)d_in[28], *bn2_b = (const float*)d_in[29];
  const float* m3_w  = (const float*)d_in[30], *m3_b  = (const float*)d_in[31];

  const int* src = ei;
  const int* dst = ei + EE;

  float* out_sig  = (float*)d_out;        // [0,8192)  sigmoid(edge_logits)
  float* out_orig = out_sig + EE;         // [8192,16384) orig_edge_logits

  // ---- workspace carve-up (deterministic every call) ----
  char* ws = (char*)d_ws;
  size_t off = 0;
  auto alloc = [&](size_t bytes) { size_t r = off; off += (bytes + 255) & ~(size_t)255; return r; };
  float*    sent   = (float*)   (ws + alloc(SS * 4));
  float*    g1     = (float*)   (ws + alloc(EE * 4));
  float*    g2     = (float*)   (ws + alloc(EE * 4));
  int*      mask   = (int*)     (ws + alloc(EE * 4));
  int*      cnt    = (int*)     (ws + alloc(256));
  _Float16* x_h    = (_Float16*)(ws + alloc((size_t)NN * FF * 2));
  _Float16* fc1wt  = (_Float16*)(ws + alloc((size_t)SS * FF * 2));       // [512,256]
  float*    h0     = (float*)   (ws + alloc((size_t)NN * SS * 4));       // fc1 C, then h0 in place
  _Float16* h0_h   = (_Float16*)(ws + alloc((size_t)NN * SS * 2));
  _Float16* c1wt   = (_Float16*)(ws + alloc((size_t)(HH*HCC) * SS * 2)); // [1024,512]
  float*    xp1    = (float*)   (ws + alloc((size_t)NN * HH * HCC * 4));
  float*    al_s1  = (float*)   (ws + alloc((size_t)NN * HH * 4));
  float*    al_d1  = (float*)   (ws + alloc((size_t)NN * HH * 4));
  float*    m1     = (float*)   (ws + alloc((size_t)NN * HH * 4));
  float*    den1   = (float*)   (ws + alloc((size_t)NN * HH * 4));
  float*    h1     = (float*)   (ws + alloc((size_t)NN * HH * HCC * 4)); // acc1 -> h1 in place
  _Float16* h1_h   = (_Float16*)(ws + alloc((size_t)NN * HH * HCC * 2));
  _Float16* c2wt   = (_Float16*)(ws + alloc((size_t)FF * (HH*HCC) * 2)); // [256,1024]
  float*    xp2    = (float*)   (ws + alloc((size_t)NN * FF * 4));
  float*    al_s2  = (float*)   (ws + alloc((size_t)NN * 4));
  float*    al_d2  = (float*)   (ws + alloc((size_t)NN * 4));
  float*    m2     = (float*)   (ws + alloc((size_t)NN * 4));
  float*    den2   = (float*)   (ws + alloc((size_t)NN * 4));
  float*    h2     = (float*)   (ws + alloc((size_t)NN * FF * 4));       // acc2 -> h2 in place
  _Float16* ef_h   = (_Float16*)(ws + alloc((size_t)EE * 4 * FF * 2));
  _Float16* m1wt   = (_Float16*)(ws + alloc((size_t)HCC * (4*FF) * 2));  // [256,1024]
  float*    z1     = (float*)   (ws + alloc((size_t)EE * HCC * 4));
  _Float16* z1_h   = (_Float16*)(ws + alloc((size_t)EE * HCC * 2));
  _Float16* m2wt   = (_Float16*)(ws + alloc((size_t)(HCC/2) * HCC * 2)); // [128,256]
  float*    z2     = (float*)   (ws + alloc((size_t)EE * (HCC/2) * 4));
  float*    bsum1  = (float*)   (ws + alloc(HCC * 4));
  float*    bsq1   = (float*)   (ws + alloc(HCC * 4));
  float*    bsum2  = (float*)   (ws + alloc((HCC/2) * 4));
  float*    bsq2   = (float*)   (ws + alloc((HCC/2) * 4));

  const int T = 256;

  // ---- weight / input conversions for WMMA ----
  k_f32_to_f16  <<<ceil_div(NN*FF, T), T, 0, stream>>>(x, x_h, NN*FF);
  k_f32_to_f16_t<<<ceil_div(FF*SS, T), T, 0, stream>>>(fc1_w, fc1wt, FF, SS);
  k_f32_to_f16_t<<<ceil_div(SS*HH*HCC, T), T, 0, stream>>>(c1_W, c1wt, SS, HH*HCC);
  k_f32_to_f16_t<<<ceil_div(HH*HCC*FF, T), T, 0, stream>>>(c2_W, c2wt, HH*HCC, FF);
  k_f32_to_f16_t<<<ceil_div(4*FF*HCC, T), T, 0, stream>>>(m1_w, m1wt, 4*FF, HCC);
  k_f32_to_f16_t<<<ceil_div(HCC*(HCC/2), T), T, 0, stream>>>(m2_w, m2wt, HCC, HCC/2);

  // ---- edge gating path (bandwidth-bound GEMVs) ----
  k_gemv_relu<<<ceil_div(SS, T), T, 0, stream>>>(sent_emb, fc0_w, fc0_b, sent, BB, SS);
  k_gemv_relu<<<ceil_div(EE, T), T, 0, stream>>>(elp, fcl_w, fcl_b, g1, EE, EE);
  k_gemv_relu<<<ceil_div(EE, T), T, 0, stream>>>(sent_emb, fce_w, fce_b, g2, BB, EE);
  k_init_f32<<<1, 64, 0, stream>>>((float*)cnt, 0.f, 64);
  k_fc2<<<ceil_div(EE, T), T, 0, stream>>>(g1, g2, fc2_w, fc2_b, out_orig, mask, cnt);

  // ---- fc1 + sent broadcast ----
  k_gemm_wmma<<<(NN/MBK)*(SS/NBK), 256, 0, stream>>>(x_h, fc1wt, h0, NN, SS, FF);
  k_fc1_epi<<<ceil_div(NN*SS, T), T, 0, stream>>>(h0, fc1_b, sent, h0_h, NN*SS);

  // ---- GAT conv1 (H=4, C=256, concat, ELU) ----
  k_gemm_wmma<<<(NN/MBK)*((HH*HCC)/NBK), 256, 0, stream>>>(h0_h, c1wt, xp1, NN, HH*HCC, SS);
  k_alpha<<<ceil_div(NN*HH, T), T, 0, stream>>>(xp1, c1_as, c1_ad, al_s1, al_d1, HH, HCC);
  k_init_f32<<<ceil_div(NN*HH, T), T, 0, stream>>>(m1, -3.0e38f, NN*HH);
  k_init_f32<<<ceil_div(NN*HH, T), T, 0, stream>>>(den1, 0.f, NN*HH);
  k_init_f32<<<ceil_div(NN*HH*HCC, T), T, 0, stream>>>(h1, 0.f, NN*HH*HCC);
  k_att_max<<<ceil_div(EE+NN, T), T, 0, stream>>>(src, dst, mask, al_s1, al_d1, m1, HH);
  k_att_agg<<<EE+NN, T, 0, stream>>>(src, dst, mask, al_s1, al_d1, m1, xp1, den1, h1, HH, HCC);
  k_att_final<<<ceil_div(NN*HH*HCC, T), T, 0, stream>>>(h1, den1, c1_b, h1_h, HH, HCC, 1);

  // ---- GAT conv2 (H=1, C=256, mean==identity, no act) ----
  k_gemm_wmma<<<(NN/MBK)*(FF/NBK), 256, 0, stream>>>(h1_h, c2wt, xp2, NN, FF, HH*HCC);
  k_alpha<<<ceil_div(NN, T), T, 0, stream>>>(xp2, c2_as, c2_ad, al_s2, al_d2, 1, FF);
  k_init_f32<<<ceil_div(NN, T), T, 0, stream>>>(m2, -3.0e38f, NN);
  k_init_f32<<<ceil_div(NN, T), T, 0, stream>>>(den2, 0.f, NN);
  k_init_f32<<<ceil_div(NN*FF, T), T, 0, stream>>>(h2, 0.f, NN*FF);
  k_att_max<<<ceil_div(EE+NN, T), T, 0, stream>>>(src, dst, mask, al_s2, al_d2, m2, 1);
  k_att_agg<<<EE+NN, T, 0, stream>>>(src, dst, mask, al_s2, al_d2, m2, xp2, den2, h2, 1, FF);
  k_att_final<<<ceil_div(NN*FF, T), T, 0, stream>>>(h2, den2, c2_b, (_Float16*)nullptr, 1, FF, 0);

  // ---- edge MLP: ef -> mlp1 -> BN -> relu -> mlp2 -> BN -> relu -> mlp3 ----
  k_ef<<<EE, FF, 0, stream>>>(h2, src, dst, ef_h);
  k_gemm_wmma<<<(EE/MBK)*(HCC/NBK), 256, 0, stream>>>(ef_h, m1wt, z1, EE, HCC, 4*FF);
  k_init_f32<<<ceil_div(HCC, T), T, 0, stream>>>(bsum1, 0.f, HCC);
  k_init_f32<<<ceil_div(HCC, T), T, 0, stream>>>(bsq1, 0.f, HCC);
  k_bn_partial<<<EE/256, HCC, 0, stream>>>(z1, mask, bsum1, bsq1, HCC);
  k_bn_apply<<<ceil_div(EE*HCC, T), T, 0, stream>>>(z1, bsum1, bsq1, cnt, bn1_g, bn1_b,
                                                    z1_h, EE*HCC, HCC);
  k_gemm_wmma<<<(EE/MBK)*((HCC/2)/NBK), 256, 0, stream>>>(z1_h, m2wt, z2, EE, HCC/2, HCC);
  k_init_f32<<<1, HCC/2, 0, stream>>>(bsum2, 0.f, HCC/2);
  k_init_f32<<<1, HCC/2, 0, stream>>>(bsq2, 0.f, HCC/2);
  k_bn_partial<<<EE/256, HCC/2, 0, stream>>>(z2, mask, bsum2, bsq2, HCC/2);
  k_bn_apply<<<ceil_div(EE*(HCC/2), T), T, 0, stream>>>(z2, bsum2, bsq2, cnt, bn2_g, bn2_b,
                                                        (_Float16*)nullptr, EE*(HCC/2), HCC/2);
  k_mlp3<<<ceil_div(EE, T), T, 0, stream>>>(z2, m3_w, m3_b, mask, out_sig);
}
